// Attention_46651934769611
// MI455X (gfx1250) — compile-verified, matrix operand order checked by reading, not run on hardware
//
#include <hip/hip_runtime.h>

#define NSEQ   4096
#define DHEAD  64
#define NHEADS 8
#define LDT    72    // LDS row stride in bf16 elems (144B = 16B-aligned rows)
#define QR     128   // Q rows per workgroup (one 16-row band per wave)

typedef __attribute__((ext_vector_type(16))) __bf16 v16bf;
typedef __attribute__((ext_vector_type(8)))  __bf16 v8bf;
typedef __attribute__((ext_vector_type(8)))  float  v8f;
typedef __attribute__((ext_vector_type(4)))  float  v4f;   // for NT vector stores

// native bf16 convert (gfx1250 has hardware bf16 cvt; RNE)
static __device__ __forceinline__ __bf16 f2bf(float f) {
    return (__bf16)f;
}
// monotone float<->uint so integer atomicMax == float max
static __device__ __forceinline__ unsigned f2ord(float f) {
    unsigned u = __builtin_bit_cast(unsigned, f);
    return (u & 0x80000000u) ? ~u : (u | 0x80000000u);
}
static __device__ __forceinline__ float ord2f(unsigned o) {
    unsigned u = (o & 0x80000000u) ? (o & 0x7fffffffu) : ~o;
    return __builtin_bit_cast(float, u);
}

// 16x32 bf16 A/B fragment from LDS per CDNA5 wave32 layout:
// lanes 0-15 : row=lane,    elems 0..7 = K[0..7],  8..15 = K[16..23]
// lanes 16-31: row=lane-16, elems 0..7 = K[8..15], 8..15 = K[24..31]
static __device__ __forceinline__ v16bf load_frag(const __bf16* base) {
    const int lane = threadIdx.x & 31;
    const __bf16* p = base + (lane & 15) * LDT + ((lane >> 4) << 3);
    v8bf lo = *(const v8bf*)(p);
    v8bf hi = *(const v8bf*)(p + 16);
    v16bf out;
#pragma unroll
    for (int i = 0; i < 8; ++i) { out[i] = lo[i]; out[8 + i] = hi[i]; }
    return out;
}

static __device__ __forceinline__ v8bf pack8(float4 a, float4 b) {
    v8bf o;
    o[0] = f2bf(a.x); o[1] = f2bf(a.y); o[2] = f2bf(a.z); o[3] = f2bf(a.w);
    o[4] = f2bf(b.x); o[5] = f2bf(b.y); o[6] = f2bf(b.z); o[7] = f2bf(b.w);
    return o;
}

// 64x64 f32 global tile -> bf16 LDS tile (packed b128 stores)
static __device__ __forceinline__ void load_tile64(const float* __restrict__ g, __bf16* s) {
    const int tid = threadIdx.x;
    const int row = tid >> 2;
    const int c0  = (tid & 3) << 4;
    const float4* gp = (const float4*)(g + row * DHEAD + c0);
    float4 f0 = gp[0], f1 = gp[1], f2 = gp[2], f3 = gp[3];
    v8bf* sp = (v8bf*)(s + row * LDT + c0);
    sp[0] = pack8(f0, f1);
    sp[1] = pack8(f2, f3);
}

// 128x64 f32 Q tile -> bf16 LDS
static __device__ __forceinline__ void load_tileQ(const float* __restrict__ g, __bf16* s) {
    const int tid = threadIdx.x;
    const int row = tid >> 1;
    const int c0  = (tid & 1) << 5;
    const float4* gp = (const float4*)(g + row * DHEAD + c0);
    v8bf* sp = (v8bf*)(s + row * LDT + c0);
#pragma unroll
    for (int i = 0; i < 4; ++i) sp[i] = pack8(gp[2 * i], gp[2 * i + 1]);
}

// 64x64 V tile transposed into LDS (d-major) so Vt A-fragments are contiguous
static __device__ __forceinline__ void load_tileVT(const float* __restrict__ g, __bf16* s) {
    const int tid = threadIdx.x;
    const int row = tid >> 2;
    const int c0  = (tid & 3) << 4;
    const float4* gp = (const float4*)(g + row * DHEAD + c0);
#pragma unroll
    for (int i = 0; i < 4; ++i) {
        float4 f = gp[i];
        const int c = c0 + 4 * i;
        s[(c + 0) * LDT + row] = f2bf(f.x);
        s[(c + 1) * LDT + row] = f2bf(f.y);
        s[(c + 2) * LDT + row] = f2bf(f.z);
        s[(c + 3) * LDT + row] = f2bf(f.w);
    }
}

// one 16x16 S^T tile: A = K-tile rows (keys), B = Q band (cols = q-rows)
static __device__ __forceinline__ v8f score_tile(const __bf16* kbase, v16bf qf0, v16bf qf1) {
    v8f c = {};
    c = __builtin_amdgcn_wmma_f32_16x16x32_bf16(false, load_frag(kbase), false, qf0,
                                                (short)0, c, false, false);
    c = __builtin_amdgcn_wmma_f32_16x16x32_bf16(false, load_frag(kbase + 32), false, qf1,
                                                (short)0, c, false, false);
    return c;
}

__global__ __launch_bounds__(256)
void attn_fused_wmma(const float* __restrict__ Q, const float* __restrict__ K,
                     const float* __restrict__ V, const int* __restrict__ Mask,
                     float* __restrict__ Out, float* __restrict__ Pout) {
    __shared__ __align__(16) __bf16 q_s [QR * LDT];
    __shared__ __align__(16) __bf16 k_s [64 * LDT];
    __shared__ __align__(16) __bf16 vt_s[64 * LDT];
    __shared__ __align__(16) float  bcol_s[64];
    __shared__ unsigned m_u[QR];
    __shared__ float    l_s[QR];

    const int tid  = threadIdx.x;
    const int lane = tid & 31;
    const int wv   = tid >> 5;            // 8 waves, one 16-row q-band each
    const int qb   = blockIdx.x * QR;
    const int bh   = blockIdx.y;
    const int b    = bh >> 3;             // H == 8

    const size_t bh_off = (size_t)bh * NSEQ;
    const int rloc = (wv << 4) + (lane & 15);  // this lane's q-row (0..127)
    const int kb8  = (lane >> 4) << 3;         // key/d sub-offset: 0 or 8

    load_tileQ(Q + (bh_off + qb) * DHEAD, q_s);
    if (tid < QR) { m_u[tid] = 0u; l_s[tid] = 0.0f; }
    const float brow = Mask[b * NSEQ + qb + rloc] ? 0.0f : -1e9f;
    __syncthreads();

    // loop-invariant Q fragments (B operand for all S^T tiles)
    const v16bf qf0 = load_frag(q_s + (wv << 4) * LDT);
    const v16bf qf1 = load_frag(q_s + (wv << 4) * LDT + 32);

    // ---------------- Pass 1: per-lane running row-max ----------------
    float mr[8];
#pragma unroll
    for (int j = 0; j < 8; ++j) mr[j] = -3.0e38f;
    for (int kc = 0; kc < NSEQ / 64; ++kc) {
        __syncthreads();
        const float* kg = K + (bh_off + (size_t)kc * 64) * DHEAD;
        load_tile64(kg, k_s);
        if (tid < 64) bcol_s[tid] = Mask[b * NSEQ + kc * 64 + tid] ? 0.0f : -1e9f;
        if (tid < 64 && kc + 1 < NSEQ / 64)
            __builtin_prefetch(kg + 64 * DHEAD + tid * DHEAD, 0, 0);
        __syncthreads();
#pragma unroll
        for (int kt = 0; kt < 4; ++kt) {
            v8f c = score_tile(k_s + (kt << 4) * LDT, qf0, qf1);
            const float4 b0 = *(const float4*)&bcol_s[(kt << 4) + kb8];
            const float4 b1 = *(const float4*)&bcol_s[(kt << 4) + kb8 + 4];
            const float bc[8] = {b0.x, b0.y, b0.z, b0.w, b1.x, b1.y, b1.z, b1.w};
#pragma unroll
            for (int j = 0; j < 8; ++j)     // independent chains per j-slot
                mr[j] = fmaxf(mr[j], fmaf(c[j], 0.125f, fminf(brow, bc[j])));
        }
    }
    float mrun = fmaxf(fmaxf(fmaxf(mr[0], mr[1]), fmaxf(mr[2], mr[3])),
                       fmaxf(fmaxf(mr[4], mr[5]), fmaxf(mr[6], mr[7])));
    atomicMax(&m_u[rloc], f2ord(mrun));   // combine the two key-halves (lanes r, r+16)
    __syncthreads();
    const float mfin = ord2f(m_u[rloc]);

    // ------- Pass 2: exp-sum (register) + O^T accumulation, P stays in VGPRs -------
    float lr[8];
#pragma unroll
    for (int j = 0; j < 8; ++j) lr[j] = 0.0f;
    v8f zacc = {};
    v8f oacc[4] = {zacc, zacc, zacc, zacc};
    for (int kc = 0; kc < NSEQ / 64; ++kc) {
        __syncthreads();
        const float* kg = K + (bh_off + (size_t)kc * 64) * DHEAD;
        const float* vg = V + (bh_off + (size_t)kc * 64) * DHEAD;
        load_tile64(kg, k_s);
        load_tileVT(vg, vt_s);
        if (tid < 64) bcol_s[tid] = Mask[b * NSEQ + kc * 64 + tid] ? 0.0f : -1e9f;
        if (tid < 64 && kc + 1 < NSEQ / 64) {
            __builtin_prefetch(kg + 64 * DHEAD + tid * DHEAD, 0, 0);
            __builtin_prefetch(vg + 64 * DHEAD + tid * DHEAD, 0, 0);
        }
        __syncthreads();

        v16bf pb[2];   // B-fragments of P^T for key chunks 0..31 / 32..63
#pragma unroll
        for (int kh = 0; kh < 2; ++kh) {
#pragma unroll
            for (int ks = 0; ks < 2; ++ks) {
                const int kt = (kh << 1) | ks;
                v8f c = score_tile(k_s + (kt << 4) * LDT, qf0, qf1);
                const float4 b0 = *(const float4*)&bcol_s[(kt << 4) + kb8];
                const float4 b1 = *(const float4*)&bcol_s[(kt << 4) + kb8 + 4];
                const float bc[8] = {b0.x, b0.y, b0.z, b0.w, b1.x, b1.y, b1.z, b1.w};
#pragma unroll
                for (int j = 0; j < 8; ++j) {
                    float p = __expf(fmaf(c[j], 0.125f, fminf(brow, bc[j])) - mfin);
                    lr[j] += p;
                    pb[kh][(ks << 3) + j] = f2bf(p);   // C-frag == B-frag layout
                }
            }
        }
#pragma unroll
        for (int db = 0; db < 4; ++db) {
#pragma unroll
            for (int kh = 0; kh < 2; ++kh) {
                v16bf a = load_frag(vt_s + (db << 4) * LDT + (kh << 5));
                oacc[db] = __builtin_amdgcn_wmma_f32_16x16x32_bf16(
                    false, a, false, pb[kh], (short)0, oacc[db], false, false);
            }
        }
    }
    float lrun = ((lr[0] + lr[1]) + (lr[2] + lr[3])) + ((lr[4] + lr[5]) + (lr[6] + lr[7]));
    atomicAdd(&l_s[rloc], lrun);
    __syncthreads();
    const float invl = 1.0f / l_s[rloc];

    // write normalized O: per-lane 8 consecutive d-values -> NT b128 stores
    {
        float* op = Out + (bh_off + qb + rloc) * DHEAD + kb8;
#pragma unroll
        for (int db = 0; db < 4; ++db) {
            v4f w0 = {oacc[db][0] * invl, oacc[db][1] * invl,
                      oacc[db][2] * invl, oacc[db][3] * invl};
            v4f w1 = {oacc[db][4] * invl, oacc[db][5] * invl,
                      oacc[db][6] * invl, oacc[db][7] * invl};
            __builtin_nontemporal_store(w0, (v4f*)(op + (db << 4)));
            __builtin_nontemporal_store(w1, (v4f*)(op + (db << 4) + 4));
        }
    }

    // ---- Pass 3: recompute S^T, stream normalized p_attn (single 1.07GB NT write) ----
    float* prow = Pout + ((size_t)(bh * NSEQ) + qb + rloc) * NSEQ + kb8;
    for (int kc = 0; kc < NSEQ / 64; ++kc) {
        __syncthreads();
        const float* kg = K + (bh_off + (size_t)kc * 64) * DHEAD;
        load_tile64(kg, k_s);
        if (tid < 64) bcol_s[tid] = Mask[b * NSEQ + kc * 64 + tid] ? 0.0f : -1e9f;
        if (tid < 64 && kc + 1 < NSEQ / 64)
            __builtin_prefetch(kg + 64 * DHEAD + tid * DHEAD, 0, 0);
        __syncthreads();
        float* pkc = prow + kc * 64;
#pragma unroll
        for (int kt = 0; kt < 4; ++kt) {
            v8f c = score_tile(k_s + (kt << 4) * LDT, qf0, qf1);
            const float4 b0 = *(const float4*)&bcol_s[(kt << 4) + kb8];
            const float4 b1 = *(const float4*)&bcol_s[(kt << 4) + kb8 + 4];
            const float bc[8] = {b0.x, b0.y, b0.z, b0.w, b1.x, b1.y, b1.z, b1.w};
            float pv[8];
#pragma unroll
            for (int j = 0; j < 8; ++j)
                pv[j] = __expf(fmaf(c[j], 0.125f, fminf(brow, bc[j])) - mfin) * invl;
            v4f w0 = {pv[0], pv[1], pv[2], pv[3]};
            v4f w1 = {pv[4], pv[5], pv[6], pv[7]};
            __builtin_nontemporal_store(w0, (v4f*)(pkc + (kt << 4)));      // keys contiguous
            __builtin_nontemporal_store(w1, (v4f*)(pkc + (kt << 4) + 4));  // per lane
        }
    }
}

extern "C" void kernel_launch(void* const* d_in, const int* in_sizes, int n_in,
                              void* d_out, int out_size, void* d_ws, size_t ws_size,
                              hipStream_t stream) {
    (void)in_sizes; (void)n_in; (void)d_ws; (void)ws_size; (void)out_size;
    const float* q    = (const float*)d_in[0];
    const float* k    = (const float*)d_in[1];
    const float* v    = (const float*)d_in[2];
    const int*   mask = (const int*)d_in[3];

    float* out = (float*)d_out;
    // tuple (out, p_attn) flat: out is B*H*N*D, then p_attn B*H*N*N
    float* p_attn = out + (size_t)2 * NHEADS * NSEQ * DHEAD;

    dim3 grid(NSEQ / QR, 2 * NHEADS);   // 32 q-bands x 16 (b,h)
    attn_fused_wmma<<<grid, 256, 0, stream>>>(q, k, v, mask, out, p_attn);
}
